// GNN_85186381349288
// MI455X (gfx1250) — compile-verified, matrix operand order checked by reading, not run on hardware
//
#include <hip/hip_runtime.h>
#include <hip/hip_bf16.h>
#include <math.h>

// ---------------------------------------------------------------------------
// Problem constants (from reference).  NOTE: gN and gE are both multiples of
// 16, so all 16-row GEMM tiles are exact and no bounds guards are needed.
// ---------------------------------------------------------------------------
constexpr int gN  = 6000;
constexpr int gE  = 12000;
constexpr int HC  = 128;
constexpr int GH  = 4;            // heads
constexpr long long KBIG = 16384; // HC*HC contraction for NNConv edge GEMM

typedef __attribute__((ext_vector_type(16))) __bf16 v16bf;
typedef __attribute__((ext_vector_type(8)))  float  v8f;

// order-preserving float<->uint mapping for atomic max
__device__ __forceinline__ unsigned int flipf(float x) {
  unsigned int u = __float_as_uint(x);
  return (u & 0x80000000u) ? ~u : (u | 0x80000000u);
}
__device__ __forceinline__ float unflipf(unsigned int u) {
  unsigned int b = (u & 0x80000000u) ? (u ^ 0x80000000u) : ~u;
  return __uint_as_float(b);
}

// Build an A fragment (ISA 16-bit A layout: lane's 16 elements are two
// contiguous runs of 8 K-values) from four float4 register quads.
__device__ __forceinline__ v16bf frag_from_f4(float4 r0, float4 r1,
                                              float4 r2, float4 r3, float s) {
  v16bf a;
  a[0]  = (__bf16)(s * r0.x); a[1]  = (__bf16)(s * r0.y);
  a[2]  = (__bf16)(s * r0.z); a[3]  = (__bf16)(s * r0.w);
  a[4]  = (__bf16)(s * r1.x); a[5]  = (__bf16)(s * r1.y);
  a[6]  = (__bf16)(s * r1.z); a[7]  = (__bf16)(s * r1.w);
  a[8]  = (__bf16)(s * r2.x); a[9]  = (__bf16)(s * r2.y);
  a[10] = (__bf16)(s * r2.z); a[11] = (__bf16)(s * r2.w);
  a[12] = (__bf16)(s * r3.x); a[13] = (__bf16)(s * r3.y);
  a[14] = (__bf16)(s * r3.z); a[15] = (__bf16)(s * r3.w);
  return a;
}

// ---------------------------------------------------------------------------
// Pack a row-major f32 B matrix [K,128] into WMMA-fragment-ready bf16 layout:
//   out[((c*8 + slab)*32 + lane)*16 + j] = B[(c*32 + kk(j,lane))*128 + slab*16 + lane%16]
// so that each lane's 16-element B fragment is one contiguous 32B read.
// ---------------------------------------------------------------------------
__global__ void pack_b(const float* __restrict__ W, __bf16* __restrict__ out,
                       int nchunks) {
  int i = blockIdx.x * blockDim.x + threadIdx.x;
  if (i >= nchunks * 4096) return;
  int c    = i >> 12;
  int r    = i & 4095;
  int slab = r >> 9;
  int lane = (r >> 4) & 31;
  int j    = r & 15;
  int kk   = j + 8 * ((j >> 3) + (lane >> 4));
  int col  = slab * 16 + (lane & 15);
  long long J = (long long)c * 32 + kk;
  out[i] = (__bf16)W[J * HC + col];
}

// ---------------------------------------------------------------------------
// tiny encoders
// ---------------------------------------------------------------------------
__global__ void encode_nodes(const float* __restrict__ x, const float* __restrict__ w,
                             float* __restrict__ h0) {
  int i = blockIdx.x * blockDim.x + threadIdx.x;
  if (i >= gN * HC) return;
  int n = i >> 7, o = i & 127;
  float s = 0.f;
#pragma unroll
  for (int k = 0; k < 6; ++k) s += x[n * 6 + k] * w[k * HC + o];
  h0[i] = s;
}

__global__ void encode_edges(const float* __restrict__ ea_in, const float* __restrict__ w,
                             const float* __restrict__ b, float* __restrict__ ea) {
  int i = blockIdx.x * blockDim.x + threadIdx.x;
  if (i >= gE * HC) return;
  int e = i >> 7, o = i & 127;
  float s = b[o];
#pragma unroll
  for (int k = 0; k < 4; ++k) s += ea_in[e * 4 + k] * w[k * HC + o];
  ea[i] = s;
}

// ---------------------------------------------------------------------------
// Dense GEMM: C[M,128] = A[M,128](f32, cvt inline) @ Bp(packed bf16 [4][8][32][16])
//             (+ bias[128]) (+ add[M,128]).  M must be a multiple of 16.
// Block = 256 threads = 8 waves; tile = 16 rows x 128 cols; wave owns 16 cols.
// ---------------------------------------------------------------------------
__global__ __launch_bounds__(256)
void dense_gemm(const float* __restrict__ A, const __bf16* __restrict__ Bp,
                const float* __restrict__ bias, const float* __restrict__ add,
                float* __restrict__ C, int M) {
  (void)M;
  int tile = blockIdx.x * 16;
  int lane = threadIdx.x & 31;
  int wave = threadIdx.x >> 5;
  int grp  = lane >> 4;
  const float* arow = A + (long long)(tile + (lane & 15)) * HC;
  v8f acc = {};
#pragma unroll
  for (int c = 0; c < 4; ++c) {
    v16bf b = *(const v16bf*)(Bp + (((c * 8 + wave) * 32 + lane) << 4));
    int kb = c * 32 + 8 * grp;
    float4 r0 = *(const float4*)(arow + kb);
    float4 r1 = *(const float4*)(arow + kb + 4);
    float4 r2 = *(const float4*)(arow + kb + 16);
    float4 r3 = *(const float4*)(arow + kb + 20);
    v16bf a = frag_from_f4(r0, r1, r2, r3, 1.0f);
    acc = __builtin_amdgcn_wmma_f32_16x16x32_bf16(false, a, false, b,
                                                  (short)0, acc, false, false);
  }
  int n = wave * 16 + (lane & 15);
  int mBase = tile + 8 * grp;
#pragma unroll
  for (int v = 0; v < 8; ++v) {
    float val = acc[v];
    if (bias) val += bias[n];
    if (add)  val += add[(long long)(mBase + v) * HC + n];
    C[(long long)(mBase + v) * HC + n] = val;
  }
}

// ---------------------------------------------------------------------------
// NNConv fused edge GEMM + segment-sum scatter.
//   msg[e,o] = sum_{J=k*128+i} (ea[e,k]*h[src[e],i]) * nn_w[J,o]
//            + sum_i h[src[e],i] * nn_b[i,o]
//   agg[dst[e], o] += msg[e,o]
// Workgroup: 8 waves, tile of 16 edges x 128 cols; K = 16384 (+128 bias term).
// A fragments are synthesized in registers from LDS (rows padded to 132 floats:
// 16B-aligned float4 reads, conflict-free banks).  B fragments come from the
// fragment-packed bf16 weights: one 32B contiguous read per lane per WMMA.
// 516 WMMAs per wave.
// ---------------------------------------------------------------------------
#define LDSP 132
__global__ __launch_bounds__(256)
void nnconv_edges(const float* __restrict__ hnodes, const float* __restrict__ ea,
                  const int* __restrict__ src, const int* __restrict__ dst,
                  const __bf16* __restrict__ Bp, const __bf16* __restrict__ Bbp,
                  float* __restrict__ agg) {
  __shared__ float ea_s[16 * LDSP];
  __shared__ float hs_s[16 * LDSP];
  int e0 = blockIdx.x * 16;
  for (int idx = threadIdx.x; idx < 16 * HC; idx += 256) {
    int m = idx >> 7, k = idx & 127;
    int e = e0 + m;
    ea_s[m * LDSP + k] = ea[(long long)e * HC + k];
    hs_s[m * LDSP + k] = hnodes[(long long)src[e] * HC + k];
  }
  __syncthreads();

  int lane = threadIdx.x & 31;
  int wave = threadIdx.x >> 5;
  int grp  = lane >> 4;
  int mrow = lane & 15;
  const float* hrow = hs_s + mrow * LDSP;
  v8f acc = {};

  for (int khi = 0; khi < HC; ++khi) {
    float eak = ea_s[mrow * LDSP + khi];
#pragma unroll
    for (int sub = 0; sub < 4; ++sub) {
      int c = khi * 4 + sub;
      const __bf16* bp = Bp + (((long long)(c * 8 + wave) * 32 + lane) << 4);
      v16bf b = *(const v16bf*)bp;
      // prefetch the B fragments two k-chunks ahead (gfx1250 global_prefetch)
      __builtin_prefetch(bp + (8LL * 32 * 16) * 2, 0, 0);
      int kb = sub * 32 + 8 * grp;
      float4 r0 = *(const float4*)(hrow + kb);
      float4 r1 = *(const float4*)(hrow + kb + 4);
      float4 r2 = *(const float4*)(hrow + kb + 16);
      float4 r3 = *(const float4*)(hrow + kb + 20);
      v16bf a = frag_from_f4(r0, r1, r2, r3, eak);
      acc = __builtin_amdgcn_wmma_f32_16x16x32_bf16(false, a, false, b,
                                                    (short)0, acc, false, false);
    }
  }
  // nn_b term: A = h_src, B = nn_b viewed as [128,128] (packed, 4 chunks)
#pragma unroll
  for (int sub = 0; sub < 4; ++sub) {
    v16bf b = *(const v16bf*)(Bbp + (((sub * 8 + wave) * 32 + lane) << 4));
    int kb = sub * 32 + 8 * grp;
    float4 r0 = *(const float4*)(hrow + kb);
    float4 r1 = *(const float4*)(hrow + kb + 4);
    float4 r2 = *(const float4*)(hrow + kb + 16);
    float4 r3 = *(const float4*)(hrow + kb + 20);
    v16bf a = frag_from_f4(r0, r1, r2, r3, 1.0f);
    acc = __builtin_amdgcn_wmma_f32_16x16x32_bf16(false, a, false, b,
                                                  (short)0, acc, false, false);
  }

  int col = wave * 16 + (lane & 15);
  int eBase = e0 + 8 * grp;
#pragma unroll
  for (int v = 0; v < 8; ++v) {
    atomicAdd(&agg[(long long)dst[eBase + v] * HC + col], acc[v]);
  }
}

// ---------------------------------------------------------------------------
// GATv2 edge kernels
// ---------------------------------------------------------------------------
__global__ void edge_logits(const float* __restrict__ xl, const float* __restrict__ xr,
                            const float* __restrict__ xe, const int* __restrict__ src,
                            const int* __restrict__ dst, const float* __restrict__ att,
                            float* __restrict__ logits, unsigned int* __restrict__ mseg) {
  int e = blockIdx.x;
  int t = threadIdx.x;              // 128 threads; wave w == head w (D==32)
  int s = src[e], d = dst[e];
  float g = xl[(long long)s * HC + t] + xr[(long long)d * HC + t] +
            xe[(long long)e * HC + t];
  g = (g > 0.f) ? g : 0.1f * g;     // leaky relu 0.1
  float p = g * att[t];
#pragma unroll
  for (int o = 16; o >= 1; o >>= 1) p += __shfl_xor(p, o, 32);
  if ((t & 31) == 0) {
    int h = t >> 5;
    logits[e * GH + h] = p;
    atomicMax(&mseg[d * GH + h], flipf(p));
  }
}

__global__ void edge_exp(float* __restrict__ logits, const int* __restrict__ dst,
                         const unsigned int* __restrict__ mseg,
                         float* __restrict__ sseg) {
  int i = blockIdx.x * blockDim.x + threadIdx.x;
  if (i >= gE * GH) return;
  int e = i >> 2, h = i & 3;
  int d = dst[e];
  float m = unflipf(mseg[d * GH + h]);
  if (!isfinite(m)) m = 0.f;
  float ex = __expf(logits[i] - m);
  logits[i] = ex;
  atomicAdd(&sseg[d * GH + h], ex);
}

__global__ void edge_aggregate(const float* __restrict__ ex, const float* __restrict__ sseg,
                               const int* __restrict__ src, const int* __restrict__ dst,
                               const float* __restrict__ xl, float* __restrict__ attout) {
  int e = blockIdx.x;
  int t = threadIdx.x; // 128
  int h = t >> 5;
  int s = src[e], d = dst[e];
  float alpha = ex[e * GH + h] / (sseg[d * GH + h] + 1e-16f);
  atomicAdd(&attout[(long long)d * HC + t], alpha * xl[(long long)s * HC + t]);
}

// ---------------------------------------------------------------------------
// LayerNorm + exact GELU; in = a (+ b) (+ bias vector)
// ---------------------------------------------------------------------------
__global__ void ln_gelu(const float* __restrict__ a, const float* __restrict__ b,
                        const float* __restrict__ bias, const float* __restrict__ g,
                        const float* __restrict__ be, float* __restrict__ out, int M) {
  (void)M;
  int row = blockIdx.x;
  int t = threadIdx.x; // 128 threads, 4 waves
  __shared__ float w1[4], w2[4];
  float v = a[(long long)row * HC + t];
  if (b)    v += b[(long long)row * HC + t];
  if (bias) v += bias[t];
  float s = v;
#pragma unroll
  for (int o = 16; o >= 1; o >>= 1) s += __shfl_xor(s, o, 32);
  if ((t & 31) == 0) w1[t >> 5] = s;
  __syncthreads();
  float mean = (w1[0] + w1[1] + w1[2] + w1[3]) * (1.f / 128.f);
  float d = v - mean;
  float q = d * d;
#pragma unroll
  for (int o = 16; o >= 1; o >>= 1) q += __shfl_xor(q, o, 32);
  if ((t & 31) == 0) w2[t >> 5] = q;
  __syncthreads();
  float var = (w2[0] + w2[1] + w2[2] + w2[3]) * (1.f / 128.f);
  float xn = d * rsqrtf(var + 1e-5f) * g[t] + be[t];
  out[(long long)row * HC + t] = 0.5f * xn * (1.f + erff(xn * 0.70710678118654752440f));
}

__global__ void add2(const float* __restrict__ a, const float* __restrict__ b,
                     float* __restrict__ o, int n) {
  int i = blockIdx.x * blockDim.x + threadIdx.x;
  int stride = gridDim.x * blockDim.x;
  for (; i < n; i += stride) o[i] = a[i] + b[i];
}

__global__ void out_proj(const float* __restrict__ t, const float* __restrict__ w,
                         const float* __restrict__ b, float* __restrict__ o, int M) {
  int n = blockIdx.x * blockDim.x + threadIdx.x;
  if (n >= M) return;
  float a0 = b[0], a1 = b[1];
#pragma unroll 4
  for (int k = 0; k < HC; ++k) {
    float v = t[(long long)n * HC + k];
    a0 += v * w[2 * k];
    a1 += v * w[2 * k + 1];
  }
  o[2 * n] = a0;
  o[2 * n + 1] = a1;
}

// ---------------------------------------------------------------------------
// Host orchestration
// ---------------------------------------------------------------------------
extern "C" void kernel_launch(void* const* d_in, const int* in_sizes, int n_in,
                              void* d_out, int out_size, void* d_ws, size_t ws_size,
                              hipStream_t stream) {
  (void)in_sizes; (void)n_in; (void)out_size; (void)ws_size;
  const float* x     = (const float*)d_in[0];
  const int*   eidx  = (const int*)d_in[1];
  const float* eattr = (const float*)d_in[2];
  const float* neww  = (const float*)d_in[4];
  const float* eew   = (const float*)d_in[5];
  const float* eeb   = (const float*)d_in[6];
  const float* wl    = (const float*)d_in[7];
  const float* bl    = (const float*)d_in[8];
  const float* wr    = (const float*)d_in[9];
  const float* br    = (const float*)d_in[10];
  const float* we    = (const float*)d_in[11];
  const float* att   = (const float*)d_in[12];
  const float* gbias = (const float*)d_in[13];
  const float* glng  = (const float*)d_in[14];
  const float* glnb  = (const float*)d_in[15];
  const float* nnw   = (const float*)d_in[16];
  const float* nnb   = (const float*)d_in[17];
  const float* root  = (const float*)d_in[18];
  const float* nnbi  = (const float*)d_in[19];
  const float* dlng  = (const float*)d_in[20];
  const float* dlnb  = (const float*)d_in[21];
  const float* outw  = (const float*)d_in[22];
  const float* outb  = (const float*)d_in[23];
  const int* src = eidx;
  const int* dst = eidx + gE;

  char* ws = (char*)d_ws;
  size_t off = 0;
  auto alloc = [&](size_t bytes) -> void* {
    void* p = ws + off;
    off = (off + bytes + 255) & ~(size_t)255;
    return p;
  };
  float* h0     = (float*)alloc((size_t)gN * HC * 4);
  float* ea     = (float*)alloc((size_t)gE * HC * 4);
  float* xl     = (float*)alloc((size_t)gN * HC * 4);
  float* xr     = (float*)alloc((size_t)gN * HC * 4);
  float* xe     = (float*)alloc((size_t)gE * HC * 4);
  float* attout = (float*)alloc((size_t)gN * HC * 4);
  float* hbuf   = (float*)alloc((size_t)gN * HC * 4);
  float* hin    = (float*)alloc((size_t)gN * HC * 4);
  float* tbuf   = (float*)alloc((size_t)gN * HC * 4);
  float* t2     = (float*)alloc((size_t)gN * HC * 4);
  float* agg    = (float*)alloc((size_t)gN * HC * 4);
  float* logits = (float*)alloc((size_t)gE * GH * 4);
  unsigned int* mseg = (unsigned int*)alloc((size_t)gN * GH * 4);
  float* sseg   = (float*)alloc((size_t)gN * GH * 4);
  __bf16* wl_bf   = (__bf16*)alloc((size_t)2 * HC * HC * 2);
  __bf16* wr_bf   = (__bf16*)alloc((size_t)2 * HC * HC * 2);
  __bf16* we_bf   = (__bf16*)alloc((size_t)2 * HC * HC * 2);
  __bf16* root_bf = (__bf16*)alloc((size_t)8 * HC * HC * 2);
  __bf16* nnw_bf  = (__bf16*)alloc((size_t)8 * KBIG * HC * 2);
  __bf16* nnb_bf  = (__bf16*)alloc((size_t)8 * KBIG * 2);

  // ---- pack all B-side weights into fragment-ready bf16 layout ----
  auto pack = [&](const float* W, __bf16* out, int nchunks) {
    int total = nchunks * 4096;
    pack_b<<<(total + 255) / 256, 256, 0, stream>>>(W, out, nchunks);
  };
  for (int i = 0; i < 2; ++i) {
    pack(wl + (size_t)i * HC * HC, wl_bf + (size_t)i * HC * HC, 4);
    pack(wr + (size_t)i * HC * HC, wr_bf + (size_t)i * HC * HC, 4);
    pack(we + (size_t)i * HC * HC, we_bf + (size_t)i * HC * HC, 4);
  }
  for (int i = 0; i < 8; ++i) {
    pack(root + (size_t)i * HC * HC, root_bf + (size_t)i * HC * HC, 4);
    pack(nnb + (size_t)i * KBIG,     nnb_bf + (size_t)i * KBIG,     4);
    pack(nnw + (size_t)i * KBIG * HC, nnw_bf + (size_t)i * KBIG * HC, 512);
  }

  // ---- encoders ----
  encode_nodes<<<(gN * HC + 255) / 256, 256, 0, stream>>>(x, neww, h0);
  encode_edges<<<(gE * HC + 255) / 256, 256, 0, stream>>>(eattr, eew, eeb, ea);

  int ngrid = gN / 16;   // exact
  int egrid = gE / 16;   // exact

  // ---- 2 GATv2 transformer layers ----
  for (int i = 0; i < 2; ++i) {
    dense_gemm<<<ngrid, 256, 0, stream>>>(h0, wl_bf + (size_t)i * HC * HC,
                                          bl + i * HC, nullptr, xl, gN);
    dense_gemm<<<ngrid, 256, 0, stream>>>(h0, wr_bf + (size_t)i * HC * HC,
                                          br + i * HC, nullptr, xr, gN);
    dense_gemm<<<egrid, 256, 0, stream>>>(ea, we_bf + (size_t)i * HC * HC,
                                          nullptr, nullptr, xe, gE);
    hipMemsetAsync(mseg, 0, (size_t)gN * GH * 4, stream);
    hipMemsetAsync(sseg, 0, (size_t)gN * GH * 4, stream);
    hipMemsetAsync(attout, 0, (size_t)gN * HC * 4, stream);
    edge_logits<<<gE, 128, 0, stream>>>(xl, xr, xe, src, dst,
                                        att + i * GH * 32, logits, mseg);
    edge_exp<<<(gE * GH + 255) / 256, 256, 0, stream>>>(logits, dst, mseg, sseg);
    edge_aggregate<<<gE, 128, 0, stream>>>(logits, sseg, src, dst, xl, attout);
    ln_gelu<<<gN, 128, 0, stream>>>(h0, attout, gbias + i * HC,
                                    glng + i * HC, glnb + i * HC, h0, gN);
  }

  // ---- initial NNConv (layer 0 weights): h = nnconv(h0) ----
  hipMemsetAsync(agg, 0, (size_t)gN * HC * 4, stream);
  nnconv_edges<<<egrid, 256, 0, stream>>>(h0, ea, src, dst,
                                          nnw_bf, nnb_bf, agg);
  dense_gemm<<<ngrid, 256, 0, stream>>>(h0, root_bf, nnbi, agg, hbuf, gN);

  // ---- 8 DeepGCN res+ layers ----
  for (int i = 0; i < 8; ++i) {
    add2<<<1024, 256, 0, stream>>>(h0, hbuf, hin, gN * HC);
    ln_gelu<<<gN, 128, 0, stream>>>(hin, nullptr, nullptr,
                                    dlng + i * HC, dlnb + i * HC, tbuf, gN);
    hipMemsetAsync(agg, 0, (size_t)gN * HC * 4, stream);
    nnconv_edges<<<egrid, 256, 0, stream>>>(tbuf, ea, src, dst,
                                            nnw_bf + (size_t)i * KBIG * HC,
                                            nnb_bf + (size_t)i * KBIG, agg);
    dense_gemm<<<ngrid, 256, 0, stream>>>(tbuf, root_bf + (size_t)i * HC * HC,
                                          nnbi + i * HC, agg, t2, gN);
    add2<<<1024, 256, 0, stream>>>(hin, t2, hbuf, gN * HC);
  }

  // ---- final LN+GELU + output projection ----
  ln_gelu<<<gN, 128, 0, stream>>>(hbuf, nullptr, nullptr, dlng, dlnb, tbuf, gN);
  out_proj<<<(gN + 127) / 128, 128, 0, stream>>>(tbuf, outw, outb,
                                                 (float*)d_out, gN);
}
